// EncoderBlock_42425686950015
// MI455X (gfx1250) — matrix operand order repaired and checked by reading, not compiled, not run on hardware
//
#include <hip/hip_runtime.h>
#include <hip/hip_bf16.h>

// ---------------------------------------------------------------------------
// CDNA5 (gfx1250) transformer encoder block.
// B=4, S=1024, D_MODEL=1024, H=16, D_K=64, D_FF=4096.
// GEMMs + attention matmuls on v_wmma_f32_16x16x32_bf16 (wave32).
// GEMM A-tiles are fetched by the Tensor Data Mover (tensor_load_to_lds),
// double-buffered and gated with s_wait_tensorcnt.
// ---------------------------------------------------------------------------

#define D_MODEL 1024
#define NHEAD   16
#define DKHEAD  64
#define D_FF    4096
#define SEQ     1024
#define BATCH   4
#define MTOK    (BATCH * SEQ)   // 4096 rows

typedef __attribute__((ext_vector_type(16))) __bf16 v16bf;
typedef __attribute__((ext_vector_type(8)))  float  v8f;
typedef __attribute__((ext_vector_type(4)))  unsigned int v4u;
typedef __attribute__((ext_vector_type(8)))  int v8i;
typedef __attribute__((ext_vector_type(4)))  int v4i;

union FragB16 { v16bf v; unsigned int u[8]; unsigned short h[16]; };
union FragF32 { v8f   v; float        f[8]; };
union G0U { v4u v; unsigned int u[4]; };
union G1U { v8i v; int i[8]; };

static __device__ inline v8f wmma_bf16(v16bf a, v16bf b, v8f c) {
  return __builtin_amdgcn_wmma_f32_16x16x32_bf16(
      false, a, false, b, (short)0, c, false, false);
}

static __device__ inline unsigned short f32_to_bf16(float x) {
  unsigned int u = __float_as_uint(x);
  unsigned int r = u + 0x7FFFu + ((u >> 16) & 1u);   // round-to-nearest-even
  return (unsigned short)(r >> 16);
}

// A-fragment (16x32 bf16) from LDS, row-major, stride `ld` ushorts.
// lanes 0-15: M=lane, K=0..15; lanes 16-31: M=lane-16, K=16..31.
static __device__ inline v16bf load_a_frag(const unsigned short* base, int ld, int lane) {
  FragB16 fr;
  int r = lane & 15, half = lane >> 4;
  const unsigned short* row = base + r * ld;
  int kb = half * 16;
#pragma unroll
  for (int j = 0; j < 4; ++j) {
    fr.u[j]     = *(const unsigned int*)(row + kb + 2 * j);
    fr.u[4 + j] = *(const unsigned int*)(row + 8 + kb + 2 * j);
  }
  return fr.v;
}

// B-fragment (32x16 bf16) from LDS stored N-major (Bt[n][k], stride `ld`).
static __device__ inline v16bf load_b_frag(const unsigned short* base, int ld, int lane) {
  FragB16 fr;
  int c = lane & 15, half = lane >> 4;
  const unsigned short* row = base + c * ld + half * 16;
#pragma unroll
  for (int j = 0; j < 8; ++j)
    fr.u[j] = *(const unsigned int*)(row + 2 * j);
  return fr.v;
}

// B-fragment from swizzled LDS: 16 contiguous ushorts per lane per 32x16
// fragment -> 2x ds_load_b128, bank-conflict free.
static __device__ inline v16bf load_b_frag_sw(const unsigned short* base, int lane) {
  FragB16 fr;
  const unsigned int* p = (const unsigned int*)(base + lane * 16);
#pragma unroll
  for (int j = 0; j < 8; ++j) fr.u[j] = p[j];
  return fr.v;
}

// ---------------------------------------------------------------------------
// TDM: load a [128 rows x 32 bf16] tile (row stride K elements) into LDS at
// `lds_off`, inserting 1 dword of padding every 16 dwords -> LDS row stride
// of 34 ushorts, matching the fragment loader layout. D# per ISA ch.8.
// ---------------------------------------------------------------------------
static __device__ inline void tdm_load_a_tile(const unsigned short* gsrc,
                                              unsigned lds_off, int K) {
  unsigned long long ga = (unsigned long long)(uintptr_t)gsrc;
  G0U g0;
  g0.u[0] = 1u;                                          // count=1 (valid user D#)
  g0.u[1] = lds_off;                                     // lds_addr (bytes)
  g0.u[2] = (unsigned)(ga & 0xFFFFFFFFu);                // global_addr[31:0]
  g0.u[3] = (unsigned)((ga >> 32) & 0x1FFFFFFu)          // global_addr[56:32]
          | (2u << 30);                                  // type=2 (image)
  G1U g1;
  g1.i[0] = (1 << 16)      // data_size = 2 bytes
          | (1 << 20)      // pad_enable
          | (3 << 22)      // pad_interval: 16 dwords
          | (0 << 25);     // pad_amount: 1 dword
  g1.i[1] = 0;                                           // tensor_dim0[15:0]=0 (1<<20)
  g1.i[2] = 16;                                          // tensor_dim0[31:16]=16 -> 1M
  g1.i[3] = 16 | (32 << 16);                             // tensor_dim1 hi; tile_dim0=32
  g1.i[4] = 128;                                         // tile_dim1=128, tile_dim2=0
  g1.i[5] = K;                                           // tensor_dim0_stride[31:0]
  g1.i[6] = 0;
  g1.i[7] = 0;
  v4i z4 = 0;
  v8i z8 = 0;
  __builtin_amdgcn_tensor_load_to_lds(g0.v, g1.v, z4, z4, z8, 0);
}

// ---------------------------------------------------------------------------
// fp32 -> bf16 elementwise (weights)
// ---------------------------------------------------------------------------
__global__ __launch_bounds__(256) void cvt_bf16_kernel(
    const float* __restrict__ in, unsigned short* __restrict__ out, int n) {
  int i = blockIdx.x * 256 + threadIdx.x;
  if (i < n) out[i] = f32_to_bf16(in[i]);
}

// ---------------------------------------------------------------------------
// LayerNorm (ddof=1, matches jnp.std) -> bf16. One block per row of 1024.
// ---------------------------------------------------------------------------
__global__ __launch_bounds__(256) void ln_bf16_kernel(
    const float* __restrict__ x, const float* __restrict__ alpha,
    const float* __restrict__ beta, unsigned short* __restrict__ out) {
  int row = blockIdx.x;
  int tid = threadIdx.x;
  const float* xr = x + (size_t)row * D_MODEL;
  float v[4], s = 0.f, s2 = 0.f;
#pragma unroll
  for (int j = 0; j < 4; ++j) {
    v[j] = xr[tid + j * 256];
    s += v[j]; s2 += v[j] * v[j];
  }
#pragma unroll
  for (int off = 16; off; off >>= 1) {
    s  += __shfl_xor(s,  off);
    s2 += __shfl_xor(s2, off);
  }
  __shared__ float rs[8], rs2[8];
  int w = tid >> 5, lane = tid & 31;
  if (lane == 0) { rs[w] = s; rs2[w] = s2; }
  __syncthreads();
  float ts = 0.f, ts2 = 0.f;
#pragma unroll
  for (int i = 0; i < 8; ++i) { ts += rs[i]; ts2 += rs2[i]; }
  float mean = ts * (1.0f / 1024.0f);
  float var  = fmaxf((ts2 - 1024.0f * mean * mean) * (1.0f / 1023.0f), 0.f);
  float inv  = alpha[0] / (sqrtf(var) + 1e-6f);
  float b0   = beta[0];
  unsigned short* orow = out + (size_t)row * D_MODEL;
#pragma unroll
  for (int j = 0; j < 4; ++j)
    orow[tid + j * 256] = f32_to_bf16((v[j] - mean) * inv + b0);
}

// ---------------------------------------------------------------------------
// bf16 GEMM: C[M,N] = A[M,K] * B[K,N] + bias (+ReLU), C in bf16 or f32.
// 128x128 block tile, BK=32, 256 threads = 8 waves, each wave 32x64.
// A tiles via TDM (double-buffered, s_wait_tensorcnt-gated); B tiles staged
// through registers into a fragment-swizzled LDS image (ds_load_b128 reads).
// ---------------------------------------------------------------------------
template <bool RELU, bool OUT_BF16>
__global__ __launch_bounds__(256) void gemm_bf16_kernel(
    const unsigned short* __restrict__ A, const unsigned short* __restrict__ B,
    const float* __restrict__ bias, float* __restrict__ Cf,
    unsigned short* __restrict__ Cb, int M, int N, int K) {
  __shared__ unsigned short As[2][128][34];   // TDM-filled, padded rows
  __shared__ unsigned short Bs[2][4096];      // swizzled: [ntile][lane][16]

  int tid = threadIdx.x, lane = tid & 31, w = tid >> 5;
  int bm = blockIdx.y * 128, bn = blockIdx.x * 128;
  int wr = w >> 1;
  int m_base = wr * 32, n_base4 = (w & 1) * 4;   // 4 n-tiles per wave

  FragF32 acc[2][4];
#pragma unroll
  for (int mt = 0; mt < 2; ++mt)
#pragma unroll
    for (int nt = 0; nt < 4; ++nt)
#pragma unroll
      for (int p = 0; p < 8; ++p) acc[mt][nt].f[p] = 0.f;

  // B staging geometry: thread t covers local n = t>>1, k half = (t&1)*16.
  int nn = tid >> 1, hh = tid & 1;
  int ks = hh * 16;
  unsigned swz = (unsigned)((nn >> 4) * 512 + ((nn & 15) + hh * 16) * 16);

  unsigned as_off[2] = { (unsigned)(uintptr_t)&As[0][0][0],
                         (unsigned)(uintptr_t)&As[1][0][0] };

  // ---- prologue: tile 0 ----
  if (w == 0) tdm_load_a_tile(A + (size_t)bm * K, as_off[0], K);
  {
    const unsigned short* src = B + (size_t)ks * N + bn + nn;
    unsigned int* dst = (unsigned int*)&Bs[0][swz];
#pragma unroll
    for (int j = 0; j < 8; ++j) {
      unsigned lo = src[(size_t)(2 * j) * N];
      unsigned hi = src[(size_t)(2 * j + 1) * N];
      dst[j] = lo | (hi << 16);
    }
  }

  int T = K >> 5;
  for (int i = 0; i < T; ++i) {
    int cur = i & 1, nxt = cur ^ 1;
    if (w == 0) __builtin_amdgcn_s_wait_tensorcnt(0);
    __syncthreads();                         // As[cur], Bs[cur] ready

    int k1 = (i + 1) << 5;
    bool have_next = (k1 < K);
    unsigned int bst[8];
    if (have_next) {
      if (w == 0) tdm_load_a_tile(A + (size_t)bm * K + k1, as_off[nxt], K);
      const unsigned short* src = B + (size_t)(k1 + ks) * N + bn + nn;
#pragma unroll
      for (int j = 0; j < 8; ++j) {
        unsigned lo = src[(size_t)(2 * j) * N];
        unsigned hi = src[(size_t)(2 * j + 1) * N];
        bst[j] = lo | (hi << 16);
      }
    }

    // ---- compute on buffer `cur` ----
    v16bf a0 = load_a_frag(&As[cur][m_base][0], 34, lane);
    v16bf a1 = load_a_frag(&As[cur][m_base + 16][0], 34, lane);
#pragma unroll
    for (int nt = 0; nt < 4; ++nt) {
      v16bf bf = load_b_frag_sw(&Bs[cur][(n_base4 + nt) * 512], lane);
      acc[0][nt].v = wmma_bf16(a0, bf, acc[0][nt].v);
      acc[1][nt].v = wmma_bf16(a1, bf, acc[1][nt].v);
    }

    if (have_next) {
      unsigned int* dst = (unsigned int*)&Bs[nxt][swz];
#pragma unroll
      for (int j = 0; j < 8; ++j) dst[j] = bst[j];
    }
    __syncthreads();                         // compute(cur) done, stage(nxt) done
  }

  int cc = lane & 15, half = lane >> 4;
#pragma unroll
  for (int mt = 0; mt < 2; ++mt)
#pragma unroll
    for (int nt = 0; nt < 4; ++nt) {
      int n = bn + (n_base4 + nt) * 16 + cc;
      float bv = bias[n];
#pragma unroll
      for (int p = 0; p < 8; ++p) {
        int m = bm + m_base + mt * 16 + half * 8 + p;
        float val = acc[mt][nt].f[p] + bv;
        if (RELU) val = fmaxf(val, 0.f);
        if (OUT_BF16) Cb[(size_t)m * N + n] = f32_to_bf16(val);
        else          Cf[(size_t)m * N + n] = val;
      }
    }
}

// ---------------------------------------------------------------------------
// Flash attention: one block = one (b,h) x 128 query rows. 8 waves, each
// owns 16 query rows. Key blocks of 64; online softmax in f32; Q*K^T and
// P*V on bf16 WMMA. ctx written bf16 in (b, q, h*64+d) layout.
// ---------------------------------------------------------------------------
__global__ __launch_bounds__(256) void attn_kernel(
    const unsigned short* __restrict__ Q, const unsigned short* __restrict__ K,
    const unsigned short* __restrict__ V, const int* __restrict__ mask,
    unsigned short* __restrict__ ctx) {
  __shared__ unsigned short Qs[128][66];   // [q][d]
  __shared__ unsigned short Kts[64][66];   // [key][d]  (B-frag for Q*K^T)
  __shared__ unsigned short Vts[64][66];   // [d][key]  (B-frag for P*V)
  __shared__ unsigned short Ps[128][66];   // [q][key]  P bf16 (A-frag)
  __shared__ int mks[64];

  int qb = blockIdx.x, bh = blockIdx.y;
  int b = bh / NHEAD, h = bh % NHEAD;
  int tid = threadIdx.x, lane = tid & 31, w = tid >> 5;
  const float NEG_INF = -__builtin_inff();

  const unsigned short* Qbase = Q + (size_t)(b * SEQ + qb * 128) * D_MODEL + h * DKHEAD;
  const unsigned short* Kbase = K + (size_t)(b * SEQ) * D_MODEL + h * DKHEAD;
  const unsigned short* Vbase = V + (size_t)(b * SEQ) * D_MODEL + h * DKHEAD;

  {   // stage Q tile 128x64
    int row = tid >> 1, cs = (tid & 1) * 32;
    const unsigned short* src = Qbase + (size_t)row * D_MODEL + cs;
#pragma unroll
    for (int j = 0; j < 16; ++j)
      *(unsigned int*)&Qs[row][cs + 2 * j] = *(const unsigned int*)(src + 2 * j);
  }

  int qr0 = w * 16;
  int cc = lane & 15, half = lane >> 4;

  float m_run[8], l_run[8];
  FragF32 o[4];
#pragma unroll
  for (int p = 0; p < 8; ++p) { m_run[p] = NEG_INF; l_run[p] = 0.f; }
#pragma unroll
  for (int nt = 0; nt < 4; ++nt)
#pragma unroll
    for (int p = 0; p < 8; ++p) o[nt].f[p] = 0.f;

  const float scale = 0.125f;   // 1/sqrt(64)

  for (int kb = 0; kb < SEQ / 64; ++kb) {
    __syncthreads();
    {   // stage K (key-major) and V (transposed) tiles of 64 keys
      int key = tid >> 2, cs = (tid & 3) * 16;
      const unsigned short* srcK = Kbase + (size_t)(kb * 64 + key) * D_MODEL + cs;
#pragma unroll
      for (int j = 0; j < 8; ++j)
        *(unsigned int*)&Kts[key][cs + 2 * j] = *(const unsigned int*)(srcK + 2 * j);
      const unsigned short* srcV = Vbase + (size_t)(kb * 64 + key) * D_MODEL + cs;
#pragma unroll
      for (int j = 0; j < 16; ++j)
        Vts[cs + j][key] = srcV[j];
    }
    if (tid < 64) mks[tid] = mask[(size_t)b * SEQ + kb * 64 + tid];
    __syncthreads();

    // ---- S = Q * K^T (16 query rows x 64 keys per wave) ----
    v16bf qa0 = load_a_frag(&Qs[qr0][0], 66, lane);
    v16bf qa1 = load_a_frag(&Qs[qr0][32], 66, lane);
    FragF32 sfr[4];
#pragma unroll
    for (int nt = 0; nt < 4; ++nt) {
      v16bf kb0 = load_b_frag(&Kts[nt * 16][0], 66, lane);
      v16bf kb1 = load_b_frag(&Kts[nt * 16][32], 66, lane);
      v8f z = {};
      sfr[nt].v = wmma_bf16(qa1, kb1, wmma_bf16(qa0, kb0, z));
    }

    // ---- scale + mask + row max ----
    float rmax[8];
#pragma unroll
    for (int p = 0; p < 8; ++p) rmax[p] = NEG_INF;
#pragma unroll
    for (int nt = 0; nt < 4; ++nt) {
      int mk = mks[nt * 16 + cc];
#pragma unroll
      for (int p = 0; p < 8; ++p) {
        float val = sfr[nt].f[p] * scale;
        if (mk == 0) val = NEG_INF;
        sfr[nt].f[p] = val;
        rmax[p] = fmaxf(rmax[p], val);
      }
    }
#pragma unroll
    for (int off = 8; off; off >>= 1)   // reduce within 16-lane halves
#pragma unroll
      for (int p = 0; p < 8; ++p)
        rmax[p] = fmaxf(rmax[p], __shfl_xor(rmax[p], off));

    // ---- online softmax update ----
    float mnew[8], fac[8], rsum[8];
#pragma unroll
    for (int p = 0; p < 8; ++p) {
      mnew[p] = fmaxf(m_run[p], rmax[p]);
      fac[p]  = (mnew[p] == NEG_INF) ? 0.f : __expf(m_run[p] - mnew[p]);
      rsum[p] = 0.f;
    }
#pragma unroll
    for (int nt = 0; nt < 4; ++nt)
#pragma unroll
      for (int p = 0; p < 8; ++p) {
        float pv = (mnew[p] == NEG_INF) ? 0.f : __expf(sfr[nt].f[p] - mnew[p]);
        sfr[nt].f[p] = pv;
        rsum[p] += pv;
      }
#pragma unroll
    for (int off = 8; off; off >>= 1)
#pragma unroll
      for (int p = 0; p < 8; ++p)
        rsum[p] += __shfl_xor(rsum[p], off);
#pragma unroll
    for (int p = 0; p < 8; ++p) {
      l_run[p] = l_run[p] * fac[p] + rsum[p];
      m_run[p] = mnew[p];
    }
#pragma unroll
    for (int nt = 0; nt < 4; ++nt)
#pragma unroll
      for (int p = 0; p < 8; ++p) o[nt].f[p] *= fac[p];

    // ---- P (C-layout) -> LDS bf16 -> A-layout; wave-private rows, and
    // same-wave LDS ops are in-order, so no barrier needed ----
#pragma unroll
    for (int nt = 0; nt < 4; ++nt)
#pragma unroll
      for (int p = 0; p < 8; ++p)
        Ps[qr0 + half * 8 + p][nt * 16 + cc] = f32_to_bf16(sfr[nt].f[p]);

    v16bf pa0 = load_a_frag(&Ps[qr0][0], 66, lane);
    v16bf pa1 = load_a_frag(&Ps[qr0][32], 66, lane);
#pragma unroll
    for (int nt = 0; nt < 4; ++nt) {
      v16bf vb0 = load_b_frag(&Vts[nt * 16][0], 66, lane);
      v16bf vb1 = load_b_frag(&Vts[nt * 16][32], 66, lane);
      o[nt].v = wmma_bf16(pa1, vb1, wmma_bf16(pa0, vb0, o[nt].v));
    }
  }

  // ---- epilogue: O / l, write ctx bf16 ----
#pragma unroll
  for (int p = 0; p < 8; ++p) {
    float inv = (l_run[p] > 0.f) ? (1.0f / l_run[p]) : 0.f;
    int qrow = qb * 128 + qr0 + half * 8 + p;
    unsigned short* dst = ctx + (size_t)(b * SEQ + qrow) * D_MODEL + h * DKHEAD;
#pragma unroll
    for (int nt = 0; nt < 4; ++nt)
      dst[nt * 16 + cc] = f32_to_bf16(o[nt].f[p] * inv);
  }
}

// ---------------------------------------------------------------------------
// Launch
// ---------------------------------------------------------------------------
extern "C" void kernel_launch(void* const* d_in, const int* in_sizes, int n_in,
                              void* d_out, int out_size, void* d_ws, size_t ws_size,
                              hipStream_t stream) {
  (void)in_sizes; (void)n_in; (void)out_size; (void)ws_size;

  const float* x    = (const float*)d_in[0];
  const int*   mask = (const int*)  d_in[1];
  const float* wq   = (const float*)d_in[2];
  const float* bq   = (const float*)d_in[3];
  const float* wk   = (const float*)d_in[4];
  const float* bk   = (const float*)d_in[5];
  const float* wv   = (const float*)d_in[6];
  const float* bv   = (const float*)d_in[7];
  const float* wo   = (const float*)d_in[8];
  const float* bo   = (const float*)d_in[9];
  const float* w1   = (const float*)d_in[10];
  const float* b1   = (const float*)d_in[11];
  const float* w2   = (const float*)d_in[12];
  const float* b2   = (const float*)d_in[13];
  const float* ln1a = (const float*)d_in[14];
  const float* ln1b = (const float*)d_in[15];
  const float* ln2a = (const float*)d_in[16];
  const float* ln2b = (const float*)d_in[17];
  float* out = (float*)d_out;

  char* ws = (char*)d_ws;
  size_t off = 0;
  auto take = [&](size_t bytes) -> void* {
    void* p = ws + off;
    off += (bytes + 255) & ~(size_t)255;
    return p;
  };

  unsigned short* wq_b = (unsigned short*)take((size_t)D_MODEL * D_MODEL * 2);
  unsigned short* wk_b = (unsigned short*)take((size_t)D_MODEL * D_MODEL * 2);
  unsigned short* wv_b = (unsigned short*)take((size_t)D_MODEL * D_MODEL * 2);
  unsigned short* wo_b = (unsigned short*)take((size_t)D_MODEL * D_MODEL * 2);
  unsigned short* w1_b = (unsigned short*)take((size_t)D_MODEL * D_FF * 2);
  unsigned short* w2_b = (unsigned short*)take((size_t)D_FF * D_MODEL * 2);
  unsigned short* n1   = (unsigned short*)take((size_t)MTOK * D_MODEL * 2);
  unsigned short* qb   = (unsigned short*)take((size_t)MTOK * D_MODEL * 2);
  unsigned short* kbuf = (unsigned short*)take((size_t)MTOK * D_MODEL * 2);
  unsigned short* vbuf = (unsigned short*)take((size_t)MTOK * D_MODEL * 2);
  unsigned short* ctxb = (unsigned short*)take((size_t)MTOK * D_MODEL * 2);
  float*          xatt = (float*)         take((size_t)MTOK * D_MODEL * 4);
  unsigned short* n2   = (unsigned short*)take((size_t)MTOK * D_MODEL * 2);
  unsigned short* h1   = (unsigned short*)take((size_t)MTOK * D_FF * 2);

  // weights -> bf16
  cvt_bf16_kernel<<<(D_MODEL * D_MODEL) / 256, 256, 0, stream>>>(wq, wq_b, D_MODEL * D_MODEL);
  cvt_bf16_kernel<<<(D_MODEL * D_MODEL) / 256, 256, 0, stream>>>(wk, wk_b, D_MODEL * D_MODEL);
  cvt_bf16_kernel<<<(D_MODEL * D_MODEL) / 256, 256, 0, stream>>>(wv, wv_b, D_MODEL * D_MODEL);
  cvt_bf16_kernel<<<(D_MODEL * D_MODEL) / 256, 256, 0, stream>>>(wo, wo_b, D_MODEL * D_MODEL);
  cvt_bf16_kernel<<<(D_MODEL * D_FF)    / 256, 256, 0, stream>>>(w1, w1_b, D_MODEL * D_FF);
  cvt_bf16_kernel<<<(D_FF * D_MODEL)    / 256, 256, 0, stream>>>(w2, w2_b, D_FF * D_MODEL);

  // LN1
  ln_bf16_kernel<<<MTOK, 256, 0, stream>>>(x, ln1a, ln1b, n1);

  // QKV projections
  dim3 gproj(D_MODEL / 128, MTOK / 128);
  gemm_bf16_kernel<false, true><<<gproj, 256, 0, stream>>>(n1, wq_b, bq, nullptr, qb,   MTOK, D_MODEL, D_MODEL);
  gemm_bf16_kernel<false, true><<<gproj, 256, 0, stream>>>(n1, wk_b, bk, nullptr, kbuf, MTOK, D_MODEL, D_MODEL);
  gemm_bf16_kernel<false, true><<<gproj, 256, 0, stream>>>(n1, wv_b, bv, nullptr, vbuf, MTOK, D_MODEL, D_MODEL);

  // attention
  attn_kernel<<<dim3(SEQ / 128, BATCH * NHEAD), 256, 0, stream>>>(qb, kbuf, vbuf, mask, ctxb);

  // output projection (f32 out, feeds LN2)
  gemm_bf16_kernel<false, false><<<gproj, 256, 0, stream>>>(ctxb, wo_b, bo, xatt, nullptr, MTOK, D_MODEL, D_MODEL);

  // LN2
  ln_bf16_kernel<<<MTOK, 256, 0, stream>>>(xatt, ln2a, ln2b, n2);

  // FFN
  gemm_bf16_kernel<true, true><<<dim3(D_FF / 128, MTOK / 128), 256, 0, stream>>>(
      n2, w1_b, b1, nullptr, h1, MTOK, D_FF, D_MODEL);
  gemm_bf16_kernel<false, false><<<dim3(D_MODEL / 128, MTOK / 128), 256, 0, stream>>>(
      h1, w2_b, b2, out, nullptr, MTOK, D_MODEL, D_FF);
}
